// MMRecBlock_29592324669442
// MI455X (gfx1250) — compile-verified, matrix operand order checked by reading, not run on hardware
//
#include <hip/hip_runtime.h>
#include <hip/hip_bf16.h>
#include <math.h>

// ---------------------------------------------------------------------------
// MMRecBlock for MI455X (gfx1250, wave32, WMMA)
//   Phase 0: pack scan weights (Wg, Wmh, Wo) to bf16 once (halves L2 traffic
//            for the 512-step scan: 24.6 GB -> 12.3 GB of weight reads)
//   Phase 1: rmsnorm + 7 bf16-WMMA GEMMs (q,k,v,z, d1, gamma, zmi)
//   Phase 2: 512 sequential steps, 3 small kernels each (stream-serialized)
//   Phase 3: rmsnorm + 2 bf16-WMMA GEMMs (FFN, fused gelu + residual)
// ---------------------------------------------------------------------------

constexpr int Bb     = 2;
constexpr int Ss     = 512;
constexpr int Dd     = 2048;
constexpr int INNERc = 512;
constexpr int Hh     = 16;
constexpr int DHc    = 128;
constexpr int FFNc   = 8192;
constexpr int SLOTSc = 32;
constexpr int Mrows  = Bb * Ss;   // 1024

typedef __attribute__((ext_vector_type(16))) __bf16          v16bf;
typedef __attribute__((ext_vector_type(16))) unsigned short  v16u;
typedef __attribute__((ext_vector_type(8)))  float           v8f;
typedef __attribute__((ext_vector_type(2)))  __bf16          v2bf;
typedef __attribute__((ext_vector_type(2)))  float           v2f;

static __device__ __forceinline__ unsigned int pack2(float lo, float hi) {
#if __has_builtin(__builtin_amdgcn_cvt_pk_bf16_f32)
  auto r = __builtin_amdgcn_cvt_pk_bf16_f32(lo, hi);
  return __builtin_bit_cast(unsigned int, r);
#else
  // fptrunc pair; backend selects native bf16 convert ops (RNE)
  v2f f = {lo, hi};
  v2bf b = __builtin_convertvector(f, v2bf);
  return __builtin_bit_cast(unsigned int, b);
#endif
}
static __device__ __forceinline__ float bflo(unsigned int u) {
  return __uint_as_float(u << 16);
}
static __device__ __forceinline__ float bfhi(unsigned int u) {
  return __uint_as_float(u & 0xFFFF0000u);
}

static __device__ __forceinline__ float apply_act(float v, int act) {
  switch (act) {
    case 1: return fmaxf(v, 0.0f);                                     // relu
    case 2: return 1.0f / (1.0f + expf(-v));                           // sigmoid
    case 3: return 0.5f * v * (1.0f + erff(v * 0.70710678118654752f)); // exact gelu
    default: return v;
  }
}

// ---------------------------------------------------------------------------
__global__ void zero_kernel(float* __restrict__ p, int n) {
  int i = blockIdx.x * blockDim.x + threadIdx.x;
  if (i < n) p[i] = 0.0f;
}

// pack fp32 pairs -> one dword of 2 bf16 (for scan GEMV weights)
__global__ void pack_bf16_kernel(const float* __restrict__ in,
                                 unsigned int* __restrict__ outp, int n2) {
  int i = blockIdx.x * blockDim.x + threadIdx.x;
  if (i < n2) outp[i] = pack2(in[2 * i], in[2 * i + 1]);
}

// ---------------------------------------------------------------------------
// RMSNorm: out = w * x / (||x||/sqrt(D) + eps) ; one 256-thread block per row
// ---------------------------------------------------------------------------
__global__ __launch_bounds__(256) void rmsnorm_kernel(
    const float* __restrict__ x, const float* __restrict__ w,
    float* __restrict__ out) {
  __shared__ float red[256];
  const size_t row = blockIdx.x;
  const float* xr = x + row * Dd;
  float s = 0.0f;
  for (int i = threadIdx.x; i < Dd; i += 256) { float v = xr[i]; s += v * v; }
  red[threadIdx.x] = s;
  __syncthreads();
  for (int st = 128; st > 0; st >>= 1) {
    if (threadIdx.x < st) red[threadIdx.x] += red[threadIdx.x + st];
    __syncthreads();
  }
  const float inv = 1.0f / (sqrtf(red[0] / (float)Dd) + 1e-6f);
  float* orow = out + row * Dd;
  for (int i = threadIdx.x; i < Dd; i += 256) orow[i] = w[i] * xr[i] * inv;
}

// ---------------------------------------------------------------------------
// bf16 WMMA GEMM: C[M,N] = act(A[M,K] @ B[K,N] + bias) (+ Res)
// 256 threads = 8 waves; block tile 128x128, wave tile 64x32, BK=32.
// Double-buffered LDS; global->reg prefetch overlaps WMMA; packed b32 LDS
// stores; __builtin_prefetch warms L2 a few k-tiles ahead.
// ---------------------------------------------------------------------------
__global__ __launch_bounds__(256) void gemm_bf16_wmma(
    const float* __restrict__ A, const float* __restrict__ Bw,
    const float* __restrict__ bias, const float* __restrict__ Res,
    float* __restrict__ C, int Ndim, int Kdim, int act) {
  const int tid  = threadIdx.x;
  const int lane = tid & 31;
  const int wid  = tid >> 5;
  const int wm   = wid >> 2;       // 0..1  (M direction)
  const int wn   = wid & 3;        // 0..3  (N direction)
  const int m0   = blockIdx.y * 128;
  const int n0   = blockIdx.x * 128;

  __shared__ unsigned short lds_a [2][128][40];  // A tile, row-major [m][k]
  __shared__ unsigned short lds_bt[2][128][40];  // B tile, transposed [n][k]

  v8f acc[4][2] = {};

  const int arow  = tid >> 3;         // 0..31
  const int acol  = (tid & 7) << 2;   // 0..28
  const int brow2 = (tid >> 5) << 1;  // 0,2,..,14 (even k rows)
  const int bcol  = (tid & 31) << 2;  // 0..124

  float4 ra[4], rb0[2], rb1[2];

  auto load_global = [&](int k0) {
#pragma unroll
    for (int p = 0; p < 4; ++p)
      ra[p] = *reinterpret_cast<const float4*>(
          &A[(size_t)(m0 + arow + p * 32) * Kdim + k0 + acol]);
#pragma unroll
    for (int p = 0; p < 2; ++p) {
      const int kk = brow2 + p * 16;
      rb0[p] = *reinterpret_cast<const float4*>(
          &Bw[(size_t)(k0 + kk) * Ndim + n0 + bcol]);
      rb1[p] = *reinterpret_cast<const float4*>(
          &Bw[(size_t)(k0 + kk + 1) * Ndim + n0 + bcol]);
    }
    if (k0 + 256 < Kdim) {  // L2 warm-up, gfx1250 global_prefetch
      __builtin_prefetch(&A[(size_t)(m0 + arow) * Kdim + k0 + 256 + acol], 0, 1);
      __builtin_prefetch(&Bw[(size_t)(k0 + 256 + brow2) * Ndim + n0 + bcol], 0, 1);
    }
  };

  auto store_lds = [&](int buf) {
#pragma unroll
    for (int p = 0; p < 4; ++p) {
      const int row = arow + p * 32;
      *reinterpret_cast<unsigned int*>(&lds_a[buf][row][acol])     = pack2(ra[p].x, ra[p].y);
      *reinterpret_cast<unsigned int*>(&lds_a[buf][row][acol + 2]) = pack2(ra[p].z, ra[p].w);
    }
#pragma unroll
    for (int p = 0; p < 2; ++p) {
      const int kk = brow2 + p * 16;  // even: pair (kk, kk+1) contiguous in [n][k]
      *reinterpret_cast<unsigned int*>(&lds_bt[buf][bcol + 0][kk]) = pack2(rb0[p].x, rb1[p].x);
      *reinterpret_cast<unsigned int*>(&lds_bt[buf][bcol + 1][kk]) = pack2(rb0[p].y, rb1[p].y);
      *reinterpret_cast<unsigned int*>(&lds_bt[buf][bcol + 2][kk]) = pack2(rb0[p].z, rb1[p].z);
      *reinterpret_cast<unsigned int*>(&lds_bt[buf][bcol + 3][kk]) = pack2(rb0[p].w, rb1[p].w);
    }
  };

  auto compute_tile = [&](int buf) {
    // Fragment layout per ISA 7.12.2 (16-bit A 16x32):
    //  lanes 0-15 : elems 0..7 = K[0..7],  elems 8..15 = K[16..23]
    //  lanes 16-31: elems 0..7 = K[8..15], elems 8..15 = K[24..31]
    const int fr   = lane & 15;
    const int koff = (lane < 16) ? 0 : 8;
    v16u af[4], bfr[2];
#pragma unroll
    for (int i = 0; i < 4; ++i) {
      const int row = wm * 64 + i * 16 + fr;
#pragma unroll
      for (int r = 0; r < 8; ++r) {
        af[i][r]     = lds_a[buf][row][koff + r];
        af[i][8 + r] = lds_a[buf][row][16 + koff + r];
      }
    }
#pragma unroll
    for (int j = 0; j < 2; ++j) {
      const int col = wn * 32 + j * 16 + fr;
#pragma unroll
      for (int r = 0; r < 8; ++r) {
        bfr[j][r]     = lds_bt[buf][col][koff + r];
        bfr[j][8 + r] = lds_bt[buf][col][16 + koff + r];
      }
    }
#pragma unroll
    for (int i = 0; i < 4; ++i)
#pragma unroll
      for (int j = 0; j < 2; ++j)
        acc[i][j] = __builtin_amdgcn_wmma_f32_16x16x32_bf16(
            false, __builtin_bit_cast(v16bf, af[i]),
            false, __builtin_bit_cast(v16bf, bfr[j]),
            (short)0, acc[i][j], false, false);
  };

  load_global(0);
  store_lds(0);
  __syncthreads();
  int buf = 0;
  for (int k0 = 32; k0 < Kdim; k0 += 32) {
    load_global(k0);        // next tile -> registers (overlaps WMMA below)
    compute_tile(buf);      // consume current LDS buffer
    store_lds(buf ^ 1);     // commit next tile to the other buffer
    __syncthreads();
    buf ^= 1;
  }
  compute_tile(buf);

  // C/D layout: VGPR r -> M = r (+8 for lanes 16-31), N = lane & 15
  const int fn    = lane & 15;
  const int rbase = (lane >> 4) << 3;  // 0 or 8
#pragma unroll
  for (int i = 0; i < 4; ++i) {
#pragma unroll
    for (int j = 0; j < 2; ++j) {
      const int col = n0 + wn * 32 + j * 16 + fn;
      const float bv = bias ? bias[col] : 0.0f;
#pragma unroll
      for (int r = 0; r < 8; ++r) {
        const int row = m0 + wm * 64 + i * 16 + rbase + r;
        float v = acc[i][j][r] + bv;
        v = apply_act(v, act);
        if (Res) v += Res[(size_t)row * Ndim + col];
        C[(size_t)row * Ndim + col] = v;
      }
    }
  }
}

// ---------------------------------------------------------------------------
// Scan step part 1: bank slot write (h_{t-1}) + gate/h_new GEMVs + state update
//   h_t = z_t*sigmoid(h@Wg+bg) + gamma_t*h + 0.1*tanh(zmi_t + h@Wmh + bm)
// Weights are pre-packed bf16 pairs; each thread produces 2 columns x 2 rows.
// grid = (D/2)/256 = 4 blocks.
// ---------------------------------------------------------------------------
__global__ __launch_bounds__(256) void scan_state_kernel(
    const unsigned int* __restrict__ Wgp, const float* __restrict__ bg,
    const unsigned int* __restrict__ Wmp, const float* __restrict__ bm,
    const float* __restrict__ z,   const float* __restrict__ zmi,
    const float* __restrict__ gam,
    const float* __restrict__ h_in, float* __restrict__ h_out,
    float* __restrict__ bank, int t) {
  __shared__ float sh[Bb * Dd];
  const int jp = blockIdx.x * blockDim.x + threadIdx.x;  // pair index 0..1023
  const int j0 = jp << 1;
  for (int i = threadIdx.x; i < Bb * Dd; i += blockDim.x) sh[i] = h_in[i];
  if (t > 0) {  // commit h_{t-1} into its circular slot before attention runs
    const int slot = (t - 1) & (SLOTSc - 1);
    *reinterpret_cast<float2*>(&bank[(size_t)slot * (Bb * Dd) + j0]) =
        *reinterpret_cast<const float2*>(&h_in[j0]);
    *reinterpret_cast<float2*>(&bank[(size_t)slot * (Bb * Dd) + Dd + j0]) =
        *reinterpret_cast<const float2*>(&h_in[Dd + j0]);
  }
  __syncthreads();

  float g00 = bg[j0], g01 = bg[j0 + 1], g10 = g00, g11 = g01;
  float m00 = bm[j0]     + zmi[(size_t)t * Dd + j0];
  float m01 = bm[j0 + 1] + zmi[(size_t)t * Dd + j0 + 1];
  float m10 = bm[j0]     + zmi[(size_t)(Ss + t) * Dd + j0];
  float m11 = bm[j0 + 1] + zmi[(size_t)(Ss + t) * Dd + j0 + 1];
#pragma unroll 4
  for (int k = 0; k < Dd; ++k) {
    const unsigned int wg2 = Wgp[(size_t)k * (Dd / 2) + jp];
    const unsigned int wm2 = Wmp[(size_t)k * (Dd / 2) + jp];
    const float h0 = sh[k], h1 = sh[Dd + k];
    g00 = fmaf(h0, bflo(wg2), g00); g01 = fmaf(h0, bfhi(wg2), g01);
    g10 = fmaf(h1, bflo(wg2), g10); g11 = fmaf(h1, bfhi(wg2), g11);
    m00 = fmaf(h0, bflo(wm2), m00); m01 = fmaf(h0, bfhi(wm2), m01);
    m10 = fmaf(h1, bflo(wm2), m10); m11 = fmaf(h1, bfhi(wm2), m11);
  }

  auto finish = [&](int b, int j, float g, float m) {
    const float gate = 1.0f / (1.0f + expf(-g));
    const float hn = tanhf(m);
    const size_t r = (size_t)(b * Ss + t) * Dd + j;
    h_out[b * Dd + j] = z[r] * gate + gam[r] * sh[b * Dd + j] + 0.1f * hn;
  };
  finish(0, j0, g00, m00); finish(0, j0 + 1, g01, m01);
  finish(1, j0, g10, m10); finish(1, j0 + 1, g11, m11);
}

// ---------------------------------------------------------------------------
// Scan step part 2: slot attention. One 128-thread block per (b,h).
// ---------------------------------------------------------------------------
__global__ __launch_bounds__(128) void attn_kernel(
    const float* __restrict__ q, const float* __restrict__ bank,
    float* __restrict__ ctx, int t) {
  const int bh = blockIdx.x;      // 0..31
  const int b = bh >> 4, h = bh & 15;
  const int d = threadIdx.x;      // 0..127
  const int wid = d >> 5, lane = d & 31;
  __shared__ float s_q[DHc];
  __shared__ float s_attn[SLOTSc];

  s_q[d] = q[((size_t)(b * Ss + t)) * Dd + h * DHc + d];
  __syncthreads();

#pragma unroll
  for (int s8 = 0; s8 < 8; ++s8) {
    const int slot = wid * 8 + s8;
    const float* kp = &bank[(size_t)slot * (Bb * Dd) + b * Dd + h * DHc];
    float p = 0.0f;
#pragma unroll
    for (int e = 0; e < 4; ++e) p += s_q[lane + 32 * e] * kp[lane + 32 * e];
    for (int off = 16; off > 0; off >>= 1) p += __shfl_xor(p, off, 32);
    if (lane == 0) s_attn[slot] = p * 0.088388347648318447f;  // 1/sqrt(128)
  }
  __syncthreads();

  if (wid == 0) {  // softmax over 32 slots in one wave
    float sc = s_attn[lane];
    float mx = sc;
    for (int off = 16; off > 0; off >>= 1) mx = fmaxf(mx, __shfl_xor(mx, off, 32));
    const float e = expf(sc - mx);
    float sm = e;
    for (int off = 16; off > 0; off >>= 1) sm += __shfl_xor(sm, off, 32);
    s_attn[lane] = e / sm;
  }
  __syncthreads();

  float acc = 0.0f;
#pragma unroll
  for (int s = 0; s < SLOTSc; ++s)
    acc += s_attn[s] * bank[(size_t)s * (Bb * Dd) + b * Dd + h * DHc + d];
  ctx[b * Dd + h * DHc + d] = acc;
}

// ---------------------------------------------------------------------------
// Scan step part 3: mem_ctx = ctx@Wo + bo ; x_res = x + h_t + mem_ctx + 0.1 v
// ---------------------------------------------------------------------------
__global__ __launch_bounds__(256) void memout_kernel(
    const unsigned int* __restrict__ Wop, const float* __restrict__ bo,
    const float* __restrict__ ctx, const float* __restrict__ h_t,
    const float* __restrict__ x, const float* __restrict__ v,
    float* __restrict__ xres, int t) {
  __shared__ float sc[Bb * Dd];
  const int jp = blockIdx.x * blockDim.x + threadIdx.x;  // pair index
  const int j0 = jp << 1;
  for (int i = threadIdx.x; i < Bb * Dd; i += blockDim.x) sc[i] = ctx[i];
  __syncthreads();
  float a00 = bo[j0], a01 = bo[j0 + 1], a10 = a00, a11 = a01;
#pragma unroll 4
  for (int k = 0; k < Dd; ++k) {
    const unsigned int w2 = Wop[(size_t)k * (Dd / 2) + jp];
    const float c0 = sc[k], c1 = sc[Dd + k];
    a00 = fmaf(c0, bflo(w2), a00); a01 = fmaf(c0, bfhi(w2), a01);
    a10 = fmaf(c1, bflo(w2), a10); a11 = fmaf(c1, bfhi(w2), a11);
  }
  auto emit = [&](int b, int j, float a) {
    const size_t r = (size_t)(b * Ss + t) * Dd + j;
    xres[r] = x[r] + h_t[b * Dd + j] + a + 0.1f * v[r];
  };
  emit(0, j0, a00); emit(0, j0 + 1, a01);
  emit(1, j0, a10); emit(1, j0 + 1, a11);
}

// ---------------------------------------------------------------------------
// Launch
// ---------------------------------------------------------------------------
extern "C" void kernel_launch(void* const* d_in, const int* in_sizes, int n_in,
                              void* d_out, int out_size, void* d_ws, size_t ws_size,
                              hipStream_t stream) {
  (void)in_sizes; (void)n_in; (void)out_size; (void)ws_size;
  const float* x   = (const float*)d_in[0];
  const float* Wq  = (const float*)d_in[1];  const float* bq  = (const float*)d_in[2];
  const float* Wk  = (const float*)d_in[3];  const float* bk  = (const float*)d_in[4];
  const float* Wv  = (const float*)d_in[5];  const float* bv  = (const float*)d_in[6];
  const float* Wz  = (const float*)d_in[7];  const float* bz  = (const float*)d_in[8];
  const float* Wg  = (const float*)d_in[9];  const float* bg  = (const float*)d_in[10];
  const float* n1w = (const float*)d_in[11]; const float* n2w = (const float*)d_in[12];
  const float* Wd1 = (const float*)d_in[13]; const float* bd1 = (const float*)d_in[14];
  const float* Wd2 = (const float*)d_in[15]; const float* bd2 = (const float*)d_in[16];
  const float* Wmi = (const float*)d_in[17];
  const float* Wmh = (const float*)d_in[18]; const float* bm  = (const float*)d_in[19];
  const float* Wo  = (const float*)d_in[20]; const float* bo  = (const float*)d_in[21];
  const float* W1f = (const float*)d_in[22]; const float* b1f = (const float*)d_in[23];
  const float* W2f = (const float*)d_in[24]; const float* b2f = (const float*)d_in[25];
  float* out = (float*)d_out;

  // workspace carve-up (4-byte units), ~131 MB total
  float* w = (float*)d_ws;
  float* xn   = w; w += (size_t)Mrows * Dd;
  float* qb   = w; w += (size_t)Mrows * Dd;
  float* kb   = w; w += (size_t)Mrows * Dd;
  float* vb   = w; w += (size_t)Mrows * Dd;
  float* zb   = w; w += (size_t)Mrows * Dd;
  float* d1   = w; w += (size_t)Mrows * INNERc;
  float* gam  = w; w += (size_t)Mrows * Dd;
  float* zmi  = w; w += (size_t)Mrows * Dd;
  float* xres = w; w += (size_t)Mrows * Dd;
  float* xn2  = w; w += (size_t)Mrows * Dd;
  float* ffn1 = w; w += (size_t)Mrows * FFNc;
  float* hA   = w; w += (size_t)Bb * Dd;
  float* hB   = w; w += (size_t)Bb * Dd;
  float* bank = w; w += (size_t)SLOTSc * Bb * Dd;
  float* ctx  = w; w += (size_t)Bb * Dd;
  unsigned int* Wgp = (unsigned int*)w; w += (size_t)Dd * Dd / 2;
  unsigned int* Wmp = (unsigned int*)w; w += (size_t)Dd * Dd / 2;
  unsigned int* Wop = (unsigned int*)w; w += (size_t)Dd * Dd / 2;

  // phase 0: state init + scan-weight packing (ws is poisoned each timing run)
  zero_kernel<<<(Bb * Dd + 255) / 256, 256, 0, stream>>>(hA, Bb * Dd);
  zero_kernel<<<(SLOTSc * Bb * Dd + 255) / 256, 256, 0, stream>>>(bank, SLOTSc * Bb * Dd);
  const int npair = Dd * Dd / 2;
  pack_bf16_kernel<<<(npair + 255) / 256, 256, 0, stream>>>(Wg,  Wgp, npair);
  pack_bf16_kernel<<<(npair + 255) / 256, 256, 0, stream>>>(Wmh, Wmp, npair);
  pack_bf16_kernel<<<(npair + 255) / 256, 256, 0, stream>>>(Wo,  Wop, npair);

  // phase 1: norm + parallel GEMMs
  rmsnorm_kernel<<<Mrows, 256, 0, stream>>>(x, n1w, xn);
  const dim3 gDD(Dd / 128, Mrows / 128);     // 16 x 8
  gemm_bf16_wmma<<<gDD, 256, 0, stream>>>(xn, Wq, bq, nullptr, qb, Dd, Dd, 0);
  gemm_bf16_wmma<<<gDD, 256, 0, stream>>>(xn, Wk, bk, nullptr, kb, Dd, Dd, 0);
  gemm_bf16_wmma<<<gDD, 256, 0, stream>>>(xn, Wv, bv, nullptr, vb, Dd, Dd, 0);
  gemm_bf16_wmma<<<gDD, 256, 0, stream>>>(xn, Wz, bz, nullptr, zb, Dd, Dd, 0);
  gemm_bf16_wmma<<<dim3(INNERc / 128, Mrows / 128), 256, 0, stream>>>(
      kb, Wd1, bd1, nullptr, d1, INNERc, Dd, 1);                       // relu
  gemm_bf16_wmma<<<gDD, 256, 0, stream>>>(d1, Wd2, bd2, nullptr, gam,
                                          Dd, INNERc, 2);              // sigmoid
  gemm_bf16_wmma<<<gDD, 256, 0, stream>>>(zb, Wmi, nullptr, nullptr, zmi, Dd, Dd, 0);

  // phase 2: sequential scan; stream order provides the per-step barrier
  float* hin = hA;
  float* hout = hB;
  for (int t = 0; t < Ss; ++t) {
    scan_state_kernel<<<Dd / 512, 256, 0, stream>>>(Wgp, bg, Wmp, bm, zb, zmi, gam,
                                                    hin, hout, bank, t);
    attn_kernel<<<Bb * Hh, 128, 0, stream>>>(qb, bank, ctx, t);
    memout_kernel<<<Dd / 512, 256, 0, stream>>>(Wop, bo, ctx, hout, x, vb, xres, t);
    float* tmp = hin; hin = hout; hout = tmp;
  }

  // phase 3: FFN
  rmsnorm_kernel<<<Mrows, 256, 0, stream>>>(xres, n2w, xn2);
  gemm_bf16_wmma<<<dim3(FFNc / 128, Mrows / 128), 256, 0, stream>>>(
      xn2, W1f, b1f, nullptr, ffn1, FFNc, Dd, 3);                      // exact gelu
  gemm_bf16_wmma<<<gDD, 256, 0, stream>>>(ffn1, W2f, b2f, xres, out, Dd, FFNc, 0);
}
